// SHaRPose_16114717294649
// MI455X (gfx1250) — matrix operand order repaired and checked by reading, not compiled, not run on hardware
//
#include <hip/hip_runtime.h>
#include <hip/hip_bf16.h>
#include <math.h>

typedef __attribute__((ext_vector_type(16))) __bf16 v16bf;
typedef __attribute__((ext_vector_type(8)))  float  v8f;
typedef unsigned int v4u __attribute__((ext_vector_type(4)));
typedef int          v8i __attribute__((ext_vector_type(8)));
typedef int          v4i __attribute__((ext_vector_type(4)));

#define DD 768
#define BB 32
#define NTOK 82
#define NPAD 96
#define NHEAD 12
#define HDIM 64
#define MROWS (BB*NTOK)   /* 2624 */
#define FROWS (BB*64)     /* 2048 */

#define EPI_BIAS 1
#define EPI_GELU 2
#define EPI_RES  4

// Hardware RNE f32->bf16 (lowers to v_fma_mixlo/hi_bf16 class ops on gfx1250)
__device__ __forceinline__ unsigned short f2bf(float f) {
  union { __bf16 h; unsigned short u; } v;
  v.h = (__bf16)f;
  return v.u;
}
__device__ __forceinline__ float bf2f(unsigned short h) {
  union { unsigned int u; float f; } v; v.u = ((unsigned int)h) << 16;
  return v.f;
}
__device__ __forceinline__ float gelu_f(float x) {
  return 0.5f * x * (1.0f + erff(x * 0.70710678118654752f));
}

union Frag {
  v16bf v;
  __bf16 h[16];
  unsigned short u[16];
  uint4 q[2];
};

// ---------------------------------------------------------------------------
// TDM 2-D tile load: DMA (tile_k x tile_rows) bf16 elements from a row-major
// tensor (row length tensor_w elems, stride stride_elems) into LDS at lds_off.
// D# layout per CDNA5 ISA 8.3/8.4 (group0 128b, group1 256b), 2-D (groups 2/3 zero).
// ---------------------------------------------------------------------------
__device__ __forceinline__ void tdm_load_2d(unsigned lds_off, const void* gaddr,
                                            unsigned tile_k, unsigned tile_rows,
                                            unsigned tensor_w, unsigned tensor_h,
                                            unsigned stride_elems)
{
  unsigned long long ga = (unsigned long long)(size_t)gaddr;
  v4u g0;
  g0[0] = 1u;                                             // count=1 (valid), no gather
  g0[1] = lds_off;                                        // lds_addr (bytes)
  g0[2] = (unsigned)(ga & 0xffffffffu);                   // global_addr[31:0]
  g0[3] = (unsigned)((ga >> 32) & 0x1ffffffu) | (2u << 30); // global_addr[56:32] | type=2
  v8i g1;
  g1[0] = (int)(1u << 16);                                // data_size=1 (2 bytes/elem)
  g1[1] = (int)((tensor_w & 0xffffu) << 16);              // tensor_dim0[15:0] @63:48
  g1[2] = (int)(((tensor_w >> 16) & 0xffffu) | ((tensor_h & 0xffffu) << 16)); // dim0 hi | dim1 lo
  g1[3] = (int)(((tensor_h >> 16) & 0xffffu) | ((tile_k & 0xffffu) << 16));   // dim1 hi | tile_dim0
  g1[4] = (int)(tile_rows & 0xffffu);                     // tile_dim1 (tile_dim2=0)
  g1[5] = (int)stride_elems;                              // tensor_dim0_stride[31:0]
  g1[6] = 0;                                              // stride hi | dim1_stride lo
  g1[7] = 0;
  v4i z4 = {0, 0, 0, 0};
  v8i z8 = {0, 0, 0, 0, 0, 0, 0, 0};
  // 6-arg toolchain variant: (g0, g1, g2, g3, extra, cpol)
  __builtin_amdgcn_tensor_load_to_lds(g0, g1, z4, z4, z8, 0);
}

// ---------------------------------------------------------------------------
// TDM-staged GEMM: C[M,N] = A(bf16)[M,K] @ W(bf16)[N,K]^T (+bias,+gelu,+res)
// 256 threads = 8 waves; block tile 128x128; wave tile 32x64 (2x4 WMMA tiles).
// Per 32-deep k-stage, wave0 issues 2 TDM DMAs (A-tile, W-tile) double-buffered
// in LDS; waves consume via ds_load_b128 fragments + 8 WMMAs.
// ---------------------------------------------------------------------------
__global__ __launch_bounds__(256) void gemm_tdm(
    const unsigned short* __restrict__ A, const unsigned short* __restrict__ W,
    const float* __restrict__ bias, const float* __restrict__ Res,
    float* __restrict__ outF, unsigned short* __restrict__ outB,
    int M, int N, int K, int flags)
{
  __shared__ __align__(16) unsigned short As[2][128][32];
  __shared__ __align__(16) unsigned short Ws[2][128][32];

  int tid  = threadIdx.x;
  int wave = tid >> 5, lane = tid & 31;
  int l    = lane & 15, half = (lane >> 4) & 1;
  int wm   = wave & 3, wn = wave >> 2;
  int blockRow = blockIdx.y * 128;
  int blockCol = blockIdx.x * 128;

  v8f acc[2][4];
#pragma unroll
  for (int i = 0; i < 2; ++i)
#pragma unroll
    for (int j = 0; j < 4; ++j) acc[i][j] = (v8f){};

  const int stages = K / 32;
  unsigned rowsA = (unsigned)(M - blockRow);
  unsigned rowsW = (unsigned)(N - blockCol);

  if (tid < 32) {  // one wave issues DMAs (EXEC ignored by TDM)
    tdm_load_2d((unsigned)(size_t)&As[0][0][0], A + (size_t)blockRow * K,
                32u, 128u, (unsigned)K, rowsA, (unsigned)K);
    tdm_load_2d((unsigned)(size_t)&Ws[0][0][0], W + (size_t)blockCol * K,
                32u, 128u, (unsigned)K, rowsW, (unsigned)K);
  }

  for (int s = 0; s < stages; ++s) {
    if (tid < 32) {
      if (s + 1 < stages) {
        int nb = (s + 1) & 1;
        int k1 = (s + 1) * 32;
        tdm_load_2d((unsigned)(size_t)&As[nb][0][0], A + (size_t)blockRow * K + k1,
                    32u, 128u, (unsigned)(K - k1), rowsA, (unsigned)K);
        tdm_load_2d((unsigned)(size_t)&Ws[nb][0][0], W + (size_t)blockCol * K + k1,
                    32u, 128u, (unsigned)(K - k1), rowsW, (unsigned)K);
        __builtin_amdgcn_s_wait_tensorcnt(2);  // stage s complete (in-order)
      } else {
        __builtin_amdgcn_s_wait_tensorcnt(0);
      }
    }
    __syncthreads();  // stage s data visible to all waves

    int cb = s & 1;
    Frag a[2], b[4];
#pragma unroll
    for (int i = 0; i < 2; ++i) {
      const unsigned short* p = &As[cb][wm * 32 + i * 16 + l][half * 16];
      a[i].q[0] = *(const uint4*)p;
      a[i].q[1] = *(const uint4*)(p + 8);
    }
#pragma unroll
    for (int j = 0; j < 4; ++j) {
      const unsigned short* p = &Ws[cb][wn * 64 + j * 16 + l][half * 16];
      b[j].q[0] = *(const uint4*)p;
      b[j].q[1] = *(const uint4*)(p + 8);
    }
#pragma unroll
    for (int i = 0; i < 2; ++i)
#pragma unroll
      for (int j = 0; j < 4; ++j)
        acc[i][j] = __builtin_amdgcn_wmma_f32_16x16x32_bf16(
            false, a[i].v, false, b[j].v, (short)0, acc[i][j], false, false);

    __syncthreads();  // all waves done with buffer cb before it is re-filled
  }

#pragma unroll
  for (int i = 0; i < 2; ++i) {
#pragma unroll
    for (int j = 0; j < 4; ++j) {
      int col = blockCol + wn * 64 + j * 16 + l;
      float bv = (flags & EPI_BIAS) ? bias[col] : 0.0f;
#pragma unroll
      for (int r = 0; r < 8; ++r) {
        int row = blockRow + wm * 32 + i * 16 + r + half * 8;
        if (row < M) {
          float v = acc[i][j][r] + bv;
          if (flags & EPI_GELU) v = gelu_f(v);
          if (flags & EPI_RES)  v += Res[(size_t)row * N + col];
          if (outF) outF[(size_t)row * N + col] = v;
          if (outB) outB[(size_t)row * N + col] = f2bf(v);
        }
      }
    }
  }
}

// ---------------------------------------------------------------------------
// Fallback GEMM (f32 weights converted in-register), direct global loads.
// ---------------------------------------------------------------------------
template<bool GUARD>
__device__ __forceinline__ void gemm_body(
    const unsigned short* __restrict__ A, const float* __restrict__ Wf,
    const float* __restrict__ bias, const float* __restrict__ Res,
    float* __restrict__ outF, unsigned short* __restrict__ outB,
    int M, int N, int K, int flags, int rowBase, int colBase, int l, int half)
{
  v8f acc[2][4];
#pragma unroll
  for (int i = 0; i < 2; ++i)
#pragma unroll
    for (int j = 0; j < 4; ++j) acc[i][j] = (v8f){};

  for (int k = 0; k < K; k += 32) {
    Frag a[2], b[4];
#pragma unroll
    for (int i = 0; i < 2; ++i) {
      int row = rowBase + i * 16 + l;
      if (!GUARD || row < M) {
        const unsigned short* p = A + (size_t)row * K + k + half * 16;
        a[i].q[0] = *(const uint4*)p;
        a[i].q[1] = *(const uint4*)(p + 8);
      } else {
        a[i].q[0] = make_uint4(0u,0u,0u,0u);
        a[i].q[1] = make_uint4(0u,0u,0u,0u);
      }
    }
#pragma unroll
    for (int j = 0; j < 4; ++j) {
      int col = colBase + j * 16 + l;
      const float* p = Wf + (size_t)col * K + k + half * 16;
#pragma unroll
      for (int e = 0; e < 16; ++e) b[j].h[e] = (__bf16)p[e];
    }
#pragma unroll
    for (int i = 0; i < 2; ++i)
#pragma unroll
      for (int j = 0; j < 4; ++j)
        acc[i][j] = __builtin_amdgcn_wmma_f32_16x16x32_bf16(
            false, a[i].v, false, b[j].v, (short)0, acc[i][j], false, false);
  }

#pragma unroll
  for (int i = 0; i < 2; ++i) {
#pragma unroll
    for (int j = 0; j < 4; ++j) {
      int col = colBase + j * 16 + l;
      float bv = (flags & EPI_BIAS) ? bias[col] : 0.0f;
#pragma unroll
      for (int r = 0; r < 8; ++r) {
        int row = rowBase + i * 16 + r + half * 8;
        if (!GUARD || row < M) {
          float v = acc[i][j][r] + bv;
          if (flags & EPI_GELU) v = gelu_f(v);
          if (flags & EPI_RES)  v += Res[(size_t)row * N + col];
          if (outF) outF[(size_t)row * N + col] = v;
          if (outB) outB[(size_t)row * N + col] = f2bf(v);
        }
      }
    }
  }
}

__global__ __launch_bounds__(256) void gemm_wmma_f32w(
    const unsigned short* __restrict__ A, const float* __restrict__ W,
    const float* __restrict__ bias, const float* __restrict__ Res,
    float* __restrict__ outF, unsigned short* __restrict__ outB,
    int M, int N, int K, int flags)
{
  int tid  = threadIdx.x;
  int wave = tid >> 5, lane = tid & 31;
  int l    = lane & 15, half = (lane >> 4) & 1;
  int wm   = wave & 3, wn = wave >> 2;
  int rowBase = blockIdx.y * 128 + wm * 32;
  int colBase = blockIdx.x * 128 + wn * 64;

  if (blockIdx.y * 128 + 128 <= M)
    gemm_body<false>(A, W, bias, Res, outF, outB, M, N, K, flags, rowBase, colBase, l, half);
  else
    gemm_body<true>(A, W, bias, Res, outF, outB, M, N, K, flags, rowBase, colBase, l, half);
}

// f32 -> bf16 bulk conversion (weights), 4 elems/thread
__global__ __launch_bounds__(256) void cvt_w_kernel(
    const float* __restrict__ s, unsigned short* __restrict__ d, size_t n)
{
  size_t i = ((size_t)blockIdx.x * 256 + threadIdx.x) * 4;
  if (i + 3 < n) {
    float4 f = *(const float4*)(s + i);
    ushort4 o;
    o.x = f2bf(f.x); o.y = f2bf(f.y); o.z = f2bf(f.z); o.w = f2bf(f.w);
    *(ushort4*)(d + i) = o;
  } else {
    for (; i < n; ++i) d[i] = f2bf(s[i]);
  }
}

// ---------------------------------------------------------------------------
// Fused attention per (b, h): S = QK^T/8, masked softmax, ga EMA, O = P V.
// 64 threads (2 waves). 82 tokens padded to 96. All matmuls via WMMA.
// ---------------------------------------------------------------------------
__global__ __launch_bounds__(64) void attn_fused(
    const unsigned short* __restrict__ qkv,  // [MROWS, 3*DD] bf16
    unsigned short* __restrict__ o,          // [MROWS, DD] bf16
    float* __restrict__ ga, int do_ema)
{
  __shared__ __align__(16) unsigned short Qs[NPAD][HDIM];
  __shared__ __align__(16) unsigned short Ks[NPAD][HDIM];
  __shared__ __align__(16) unsigned short Vt[HDIM][NPAD];
  __shared__ __align__(16) unsigned short Ps[NPAD][NPAD];  // logits then probs (bf16)

  int tid = threadIdx.x;
  int b = blockIdx.x / NHEAD, h = blockIdx.x % NHEAD;

  for (int idx = tid; idx < NPAD * HDIM; idx += 64) {
    int t = idx / HDIM, d = idx % HDIM;
    unsigned short q = 0, kk = 0, vv = 0;
    if (t < NTOK) {
      const unsigned short* base = qkv + (size_t)(b * NTOK + t) * (3 * DD) + h * HDIM + d;
      q  = base[0];
      kk = base[DD];
      vv = base[2 * DD];
    }
    Qs[t][d] = q; Ks[t][d] = kk; Vt[d][t] = vv;
  }
  __syncthreads();

  int wave = tid >> 5, lane = tid & 31;
  int l = lane & 15, half = lane >> 4;
  const float scale = 0.125f;  // HD^-0.5

  // S = Q K^T
  for (int rt = wave * 3; rt < wave * 3 + 3; ++rt) {
    for (int ct = 0; ct < 6; ++ct) {
      v8f acc = {};
#pragma unroll
      for (int ks = 0; ks < HDIM; ks += 32) {
        Frag a, bf;
        const unsigned short* pa = &Qs[rt * 16 + l][ks + half * 16];
        a.q[0] = *(const uint4*)pa;
        a.q[1] = *(const uint4*)(pa + 8);
        const unsigned short* pb = &Ks[ct * 16 + l][ks + half * 16];
        bf.q[0] = *(const uint4*)pb;
        bf.q[1] = *(const uint4*)(pb + 8);
        acc = __builtin_amdgcn_wmma_f32_16x16x32_bf16(false, a.v, false, bf.v, (short)0, acc, false, false);
      }
      int col = ct * 16 + l;
#pragma unroll
      for (int r = 0; r < 8; ++r) {
        int row = rt * 16 + r + half * 8;
        Ps[row][col] = f2bf((col < NTOK) ? acc[r] * scale : -1e30f);
      }
    }
  }
  __syncthreads();

  // softmax rows (+ ga EMA)
  for (int r = tid; r < NPAD; r += 64) {
    float m = -1e30f;
    for (int c = 0; c < NPAD; ++c) m = fmaxf(m, bf2f(Ps[r][c]));
    float s = 0.f;
    for (int c = 0; c < NPAD; ++c) s += __expf(bf2f(Ps[r][c]) - m);
    float inv = 1.0f / s;
    for (int c = 0; c < NPAD; ++c) {
      float p = __expf(bf2f(Ps[r][c]) - m) * inv;
      Ps[r][c] = f2bf(p);
      if (do_ema && r < NTOK && c < NTOK) {
        size_t gi = ((size_t)(b * NHEAD + h) * NTOK + r) * NTOK + c;
        ga[gi] = 0.99f * ga[gi] + 0.01f * p;
      }
    }
  }
  __syncthreads();

  // O = P V
  for (int rt = wave * 3; rt < wave * 3 + 3; ++rt) {
    for (int ct = 0; ct < 4; ++ct) {
      v8f acc = {};
#pragma unroll
      for (int ks = 0; ks < NPAD; ks += 32) {
        Frag a, bf;
        const unsigned short* pa = &Ps[rt * 16 + l][ks + half * 16];
        a.q[0] = *(const uint4*)pa;
        a.q[1] = *(const uint4*)(pa + 8);
        const unsigned short* pb = &Vt[ct * 16 + l][ks + half * 16];
        bf.q[0] = *(const uint4*)pb;
        bf.q[1] = *(const uint4*)(pb + 8);
        acc = __builtin_amdgcn_wmma_f32_16x16x32_bf16(false, a.v, false, bf.v, (short)0, acc, false, false);
      }
      int col = ct * 16 + l;
#pragma unroll
      for (int r = 0; r < 8; ++r) {
        int row = rt * 16 + r + half * 8;
        if (row < NTOK)
          o[(size_t)(b * NTOK + row) * DD + h * HDIM + col] = f2bf(acc[r]);
      }
    }
  }
}

// ---------------------------------------------------------------------------
// LayerNorm: one wave per row of 768; outputs bf16 and/or fp32.
// ---------------------------------------------------------------------------
__global__ __launch_bounds__(256) void ln_kernel(
    const float* __restrict__ x, const float* __restrict__ w, const float* __restrict__ b,
    unsigned short* __restrict__ outB, float* __restrict__ outF, int M)
{
  int row = blockIdx.x * 8 + (threadIdx.x >> 5);
  int lane = threadIdx.x & 31;
  if (row >= M) return;
  const float* xr = x + (size_t)row * DD;
  float r[24];
  float s = 0.f;
#pragma unroll
  for (int i = 0; i < 24; ++i) { r[i] = xr[lane + i * 32]; s += r[i]; }
#pragma unroll
  for (int off = 16; off >= 1; off >>= 1) s += __shfl_xor(s, off, 32);
  float mu = s * (1.0f / 768.0f);
  float var = 0.f;
#pragma unroll
  for (int i = 0; i < 24; ++i) { float d = r[i] - mu; var += d * d; }
#pragma unroll
  for (int off = 16; off >= 1; off >>= 1) var += __shfl_xor(var, off, 32);
  float inv = rsqrtf(var * (1.0f / 768.0f) + 1e-5f);
#pragma unroll
  for (int i = 0; i < 24; ++i) {
    int c = lane + i * 32;
    float y = (r[i] - mu) * inv * w[c] + b[c];
    if (outB) outB[(size_t)row * DD + c] = f2bf(y);
    if (outF) outF[(size_t)row * DD + c] = y;
  }
}

// bilinear 256->128 (2x2 average) + patch extraction -> bf16 [FROWS, 768]
__global__ void patchify_kernel(const float* __restrict__ img, unsigned short* __restrict__ patches)
{
  int idx = blockIdx.x * blockDim.x + threadIdx.x;
  if (idx >= FROWS * DD) return;
  int row = idx / DD, col = idx % DD;
  int bb = row / 64, tok = row % 64;
  int h0 = tok / 8, w0 = tok % 8;
  int c = col / 256, rem = col % 256;
  int py = rem / 16, px = rem % 16;
  int y = (h0 * 16 + py) * 2, x = (w0 * 16 + px) * 2;
  const float* p = img + (((size_t)bb * 3 + c) * 256 + y) * 256 + x;
  patches[idx] = f2bf(0.25f * (p[0] + p[1] + p[256] + p[257]));
}

__global__ void build_x_kernel(const float* __restrict__ pe, const float* __restrict__ pos,
                               const float* __restrict__ qtok, const float* __restrict__ kp,
                               float* __restrict__ x)
{
  int idx = blockIdx.x * blockDim.x + threadIdx.x;
  if (idx >= MROWS * DD) return;
  int m = idx / DD, d = idx % DD;
  int bb = m / NTOK, t = m % NTOK;
  float v;
  if (t == 0)       v = qtok[d];
  else if (t <= 17) v = kp[(t - 1) * DD + d];
  else { int pt = t - 18; v = pe[(size_t)(bb * 64 + pt) * DD + d] + pos[pt * DD + d]; }
  x[idx] = v;
}

__global__ void gather_tokens_kernel(const float* __restrict__ xn, float* __restrict__ dst,
                                     int tokStart, int tokCount)
{
  int idx = blockIdx.x * blockDim.x + threadIdx.x;
  int total = BB * tokCount * DD;
  if (idx >= total) return;
  int d = idx % DD; int r = idx / DD;
  int bb = r / tokCount; int t = r % tokCount;
  dst[idx] = xn[((size_t)bb * NTOK + tokStart + t) * DD + d];
}

__global__ void upsample_kernel(const float* __restrict__ feat, float* __restrict__ outp)
{
  int idx = blockIdx.x * blockDim.x + threadIdx.x;
  if (idx >= BB * 256 * DD) return;
  int d = idx % DD; int r = idx / DD;
  int bb = r / 256; int t = r % 256;
  int y = t / 16, x = t % 16;
  int t0 = (y >> 1) * 8 + (x >> 1);
  outp[idx] = feat[((size_t)bb * 64 + t0) * DD + d];
}

__global__ __launch_bounds__(32) void qual_final_kernel(
    const unsigned short* __restrict__ q2, const float* __restrict__ w3,
    const float* __restrict__ b3, float* __restrict__ outq)
{
  int bb = blockIdx.x; int lane = threadIdx.x;
  float s = 0.f;
  for (int i = lane; i < DD; i += 32) s += bf2f(q2[bb * DD + i]) * w3[i];
  for (int off = 16; off >= 1; off >>= 1) s += __shfl_xor(s, off, 32);
  if (lane == 0) outq[bb] = gelu_f(s + b3[0]);
}

// ---------------------------------------------------------------------------
extern "C" void kernel_launch(void* const* d_in, const int* in_sizes, int n_in,
                              void* d_out, int out_size, void* d_ws, size_t ws_size,
                              hipStream_t stream)
{
  (void)in_sizes; (void)n_in; (void)out_size;
  const float* img     = (const float*)d_in[0];
  const float* kp      = (const float*)d_in[1];
  const float* qtok    = (const float*)d_in[2];
  const float* pos0    = (const float*)d_in[3];
  const float* patch_w = (const float*)d_in[4];
  const float* patch_b = (const float*)d_in[5];
  const float* ln1_w   = (const float*)d_in[6];
  const float* ln1_b   = (const float*)d_in[7];
  const float* qkv_w   = (const float*)d_in[8];
  const float* proj_w  = (const float*)d_in[9];
  const float* proj_b  = (const float*)d_in[10];
  const float* ln2_w   = (const float*)d_in[11];
  const float* ln2_b   = (const float*)d_in[12];
  const float* fc1_w   = (const float*)d_in[13];
  const float* fc1_b   = (const float*)d_in[14];
  const float* fc2_w   = (const float*)d_in[15];
  const float* fc2_b   = (const float*)d_in[16];
  const float* norm_w  = (const float*)d_in[17];
  const float* norm_b  = (const float*)d_in[18];
  const float* r_ln_w  = (const float*)d_in[19];
  const float* r_ln_b  = (const float*)d_in[20];
  const float* r_fc1_w = (const float*)d_in[21];
  const float* r_fc1_b = (const float*)d_in[22];
  const float* r_fc2_w = (const float*)d_in[23];
  const float* r_fc2_b = (const float*)d_in[24];
  const float* qp_norm_w = (const float*)d_in[25];
  const float* qp_norm_b = (const float*)d_in[26];
  const float* qp_l1_w = (const float*)d_in[27];
  const float* qp_l1_b = (const float*)d_in[28];
  const float* qp_l2_w = (const float*)d_in[29];
  const float* qp_l2_b = (const float*)d_in[30];
  const float* qp_l3_w = (const float*)d_in[31];
  const float* qp_l3_b = (const float*)d_in[32];

  char* ws = (char*)d_ws;
  size_t off = 0;
  auto alloc = [&](size_t bytes) {
    void* p = ws + off;
    off = (off + bytes + 255) & ~(size_t)255;
    return p;
  };

  // activations
  float*          pe      = (float*)         alloc((size_t)FROWS * DD * 4);
  unsigned short* patches = (unsigned short*)alloc((size_t)FROWS * DD * 2);
  float*          x       = (float*)         alloc((size_t)MROWS * DD * 4);
  unsigned short* hbuf    = (unsigned short*)alloc((size_t)MROWS * DD * 2);
  unsigned short* qkvb    = (unsigned short*)alloc((size_t)MROWS * 3 * DD * 2);
  unsigned short* obuf    = (unsigned short*)alloc((size_t)MROWS * DD * 2);
  unsigned short* h3      = (unsigned short*)alloc((size_t)MROWS * 4 * DD * 2);
  float*          xn      = (float*)         alloc((size_t)MROWS * DD * 4);
  float*          qrow    = (float*)         alloc((size_t)BB * DD * 4);
  unsigned short* qh      = (unsigned short*)alloc((size_t)BB * DD * 2);
  unsigned short* q1b     = (unsigned short*)alloc((size_t)BB * DD * 2);
  unsigned short* q2b     = (unsigned short*)alloc((size_t)BB * DD * 2);
  float*          fbuf    = (float*)         alloc((size_t)FROWS * DD * 4);
  unsigned short* fh      = (unsigned short*)alloc((size_t)FROWS * DD * 2);
  unsigned short* rf1     = (unsigned short*)alloc((size_t)FROWS * 4 * DD * 2);
  float*          featb   = (float*)         alloc((size_t)FROWS * DD * 4);

  // bf16 weight staging (used only if d_ws is large enough)
  const size_t n_patchw = (size_t)DD * DD;
  const size_t n_qkvw   = (size_t)12 * 3 * DD * DD;
  const size_t n_projw  = (size_t)12 * DD * DD;
  const size_t n_fc1w   = (size_t)12 * 4 * DD * DD;
  const size_t n_fc2w   = (size_t)12 * 4 * DD * DD;
  const size_t n_rfc1w  = (size_t)4 * DD * DD;
  const size_t n_rfc2w  = (size_t)4 * DD * DD;
  const size_t n_qplw   = (size_t)DD * DD;
  unsigned short* patch_wb = (unsigned short*)alloc(n_patchw * 2);
  unsigned short* qkv_wb   = (unsigned short*)alloc(n_qkvw * 2);
  unsigned short* proj_wb  = (unsigned short*)alloc(n_projw * 2);
  unsigned short* fc1_wb   = (unsigned short*)alloc(n_fc1w * 2);
  unsigned short* fc2_wb   = (unsigned short*)alloc(n_fc2w * 2);
  unsigned short* rfc1_wb  = (unsigned short*)alloc(n_rfc1w * 2);
  unsigned short* rfc2_wb  = (unsigned short*)alloc(n_rfc2w * 2);
  unsigned short* qpl1_wb  = (unsigned short*)alloc(n_qplw * 2);
  unsigned short* qpl2_wb  = (unsigned short*)alloc(n_qplw * 2);
  const bool preconv = (off <= ws_size);

  float* out_kpt = (float*)d_out;                       // [32,17,768]
  float* out_q   = out_kpt + (size_t)BB * 17 * DD;      // [32,1]
  float* out_ff  = out_q + BB;                          // [32,256,768]
  float* out_ga  = out_ff + (size_t)BB * 256 * DD;      // [32,12,82,82]

  (void)hipMemsetAsync(out_ga, 0, (size_t)BB * NHEAD * NTOK * NTOK * 4, stream);

  auto cvt = [&](const float* s, unsigned short* d, size_t n) {
    cvt_w_kernel<<<(unsigned)((n / 4 + 255) / 256), 256, 0, stream>>>(s, d, n);
  };
  if (preconv) {
    cvt(patch_w, patch_wb, n_patchw);
    cvt(qkv_w,   qkv_wb,   n_qkvw);
    cvt(proj_w,  proj_wb,  n_projw);
    cvt(fc1_w,   fc1_wb,   n_fc1w);
    cvt(fc2_w,   fc2_wb,   n_fc2w);
    cvt(r_fc1_w, rfc1_wb,  n_rfc1w);
    cvt(r_fc2_w, rfc2_wb,  n_rfc2w);
    cvt(qp_l1_w, qpl1_wb,  n_qplw);
    cvt(qp_l2_w, qpl2_wb,  n_qplw);
  }

  auto gemm = [&](const unsigned short* A, const float* Wf, const unsigned short* Wb,
                  const float* bias, const float* Res, float* oF, unsigned short* oB,
                  int M, int N, int K, int flags) {
    dim3 grid(N / 128, (M + 127) / 128);
    if (preconv)
      gemm_tdm<<<grid, 256, 0, stream>>>(A, Wb, bias, Res, oF, oB, M, N, K, flags);
    else
      gemm_wmma_f32w<<<grid, 256, 0, stream>>>(A, Wf, bias, Res, oF, oB, M, N, K, flags);
  };

  patchify_kernel<<<(FROWS * DD + 255) / 256, 256, 0, stream>>>(img, patches);
  gemm(patches, patch_w, patch_wb, patch_b, nullptr, pe, nullptr, FROWS, DD, DD, EPI_BIAS);
  build_x_kernel<<<(MROWS * DD + 255) / 256, 256, 0, stream>>>(pe, pos0, qtok, kp, x);

  for (int lyr = 0; lyr < 12; ++lyr) {
    const size_t wo1 = (size_t)lyr * 3 * DD * DD;
    const size_t wo2 = (size_t)lyr * DD * DD;
    const size_t wo3 = (size_t)lyr * 4 * DD * DD;
    ln_kernel<<<(MROWS + 7) / 8, 256, 0, stream>>>(x, ln1_w + lyr * DD, ln1_b + lyr * DD, hbuf, nullptr, MROWS);
    gemm(hbuf, qkv_w + wo1, qkv_wb + wo1, nullptr, nullptr, nullptr, qkvb, MROWS, 3 * DD, DD, 0);
    attn_fused<<<BB * NHEAD, 64, 0, stream>>>(qkvb, obuf, out_ga, (lyr >= 3) ? 1 : 0);
    gemm(obuf, proj_w + wo2, proj_wb + wo2, proj_b + lyr * DD, x, x, nullptr, MROWS, DD, DD, EPI_BIAS | EPI_RES);
    ln_kernel<<<(MROWS + 7) / 8, 256, 0, stream>>>(x, ln2_w + lyr * DD, ln2_b + lyr * DD, hbuf, nullptr, MROWS);
    gemm(hbuf, fc1_w + wo3, fc1_wb + wo3, fc1_b + lyr * 4 * DD, nullptr, nullptr, h3, MROWS, 4 * DD, DD, EPI_BIAS | EPI_GELU);
    gemm(h3, fc2_w + wo3, fc2_wb + wo3, fc2_b + lyr * DD, x, x, nullptr, MROWS, DD, 4 * DD, EPI_BIAS | EPI_RES);
  }

  ln_kernel<<<(MROWS + 7) / 8, 256, 0, stream>>>(x, norm_w, norm_b, nullptr, xn, MROWS);

  // keypoint tokens out
  gather_tokens_kernel<<<(BB * 17 * DD + 255) / 256, 256, 0, stream>>>(xn, out_kpt, 1, 17);

  // quality head
  gather_tokens_kernel<<<(BB * DD + 255) / 256, 256, 0, stream>>>(xn, qrow, 0, 1);
  ln_kernel<<<(BB + 7) / 8, 256, 0, stream>>>(qrow, qp_norm_w, qp_norm_b, qh, nullptr, BB);
  gemm(qh, qp_l1_w, qpl1_wb, qp_l1_b, nullptr, nullptr, q1b, BB, DD, DD, EPI_BIAS | EPI_GELU);
  gemm(q1b, qp_l2_w, qpl2_wb, qp_l2_b, nullptr, nullptr, q2b, BB, DD, DD, EPI_BIAS | EPI_GELU);
  qual_final_kernel<<<BB, 32, 0, stream>>>(q2b, qp_l3_w, qp_l3_b, out_q);

  // refine head + 2x upsample
  gather_tokens_kernel<<<(FROWS * DD + 255) / 256, 256, 0, stream>>>(xn, fbuf, 18, 64);
  ln_kernel<<<(FROWS + 7) / 8, 256, 0, stream>>>(fbuf, r_ln_w, r_ln_b, fh, nullptr, FROWS);
  gemm(fh, r_fc1_w, rfc1_wb, r_fc1_b, nullptr, nullptr, rf1, FROWS, 4 * DD, DD, EPI_BIAS | EPI_GELU);
  gemm(rf1, r_fc2_w, rfc2_wb, r_fc2_b, nullptr, featb, nullptr, FROWS, DD, 4 * DD, EPI_BIAS);
  upsample_kernel<<<(BB * 256 * DD + 255) / 256, 256, 0, stream>>>(featb, out_ff);
}
